// PointNetFPBlock_43413529428270
// MI455X (gfx1250) — matrix operand order repaired and checked by reading, not compiled, not run on hardware
//
#include <hip/hip_runtime.h>

typedef unsigned short u16;
typedef __bf16 bf16_t;
typedef bf16_t v16bf __attribute__((ext_vector_type(16)));
typedef float v8f  __attribute__((ext_vector_type(8)));
typedef float v4f  __attribute__((ext_vector_type(4)));
typedef unsigned int v4u __attribute__((ext_vector_type(4)));

#define B_  16
#define N1_ 1024
#define N2_ 4096
#define C_  512
#define M_TOTAL (B_ * N2_)   /* 65536 */
#define HSTRIDE 520          /* 512 + 8 bf16 pad: 1040 B/row -> +4 banks per row */
#define LDS_BYTES (128 * HSTRIDE * 2)

__device__ __forceinline__ u16 f2bf(float f) {
  unsigned int u = __float_as_uint(f);
  u += 0x7fffu + ((u >> 16) & 1u);   // round-to-nearest-even
  return (u16)(u >> 16);
}

// ---------------------------------------------------------------- weights f32 -> bf16
__global__ __launch_bounds__(256) void cvt_bf16_kernel(const float* __restrict__ src,
                                                       u16* __restrict__ dst, int n) {
  int i = blockIdx.x * 256 + threadIdx.x;
  if (i < n) dst[i] = f2bf(src[i]);
}

// ---------------------------------------------------------------- 3-NN interpolation
// one wave32 per dense point; 8 waves (points) per block, all in the same batch.
__global__ __launch_bounds__(256) void interp_kernel(const float* __restrict__ xyzd,
                                                     const float* __restrict__ xyzs,
                                                     const float* __restrict__ feat,
                                                     u16* __restrict__ X) {
  __shared__ float sxyz[N1_ * 3];      // 12 KB: sparse coords for this batch
  __shared__ float sdist[8][96];
  __shared__ int   sidx[8][96];

  const int lane = threadIdx.x & 31;
  const int wave = threadIdx.x >> 5;
  const int ptBase = blockIdx.x * 8;
  const int b = ptBase / N2_;          // uniform for the whole block (4096 % 8 == 0)

  const float* xs = xyzs + (size_t)b * N1_ * 3;
  for (int t = threadIdx.x; t < N1_ * 3; t += 256) sxyz[t] = xs[t];
  __syncthreads();

  const int pt = ptBase + wave;
  const float* xd = xyzd + (size_t)pt * 3;
  const float px = xd[0], py = xd[1], pz = xd[2];

  // per-lane top-3 over strided candidates
  float bd0 = 3.4e38f, bd1 = 3.4e38f, bd2 = 3.4e38f;
  int   bi0 = 0, bi1 = 0, bi2 = 0;
  for (int s = lane; s < N1_; s += 32) {
    float dx = px - sxyz[3 * s + 0];
    float dy = py - sxyz[3 * s + 1];
    float dz = pz - sxyz[3 * s + 2];
    float d2 = dx * dx + dy * dy + dz * dz;
    if (d2 < bd0)      { bd2 = bd1; bi2 = bi1; bd1 = bd0; bi1 = bi0; bd0 = d2; bi0 = s; }
    else if (d2 < bd1) { bd2 = bd1; bi2 = bi1; bd1 = d2;  bi1 = s; }
    else if (d2 < bd2) { bd2 = d2;  bi2 = s; }
  }
  sdist[wave][lane * 3 + 0] = bd0; sidx[wave][lane * 3 + 0] = bi0;
  sdist[wave][lane * 3 + 1] = bd1; sidx[wave][lane * 3 + 1] = bi1;
  sdist[wave][lane * 3 + 2] = bd2; sidx[wave][lane * 3 + 2] = bi2;
  __syncthreads();

  // every lane redundantly reduces the 96 candidates (LDS broadcast reads)
  float fd0 = 3.4e38f, fd1 = 3.4e38f, fd2 = 3.4e38f;
  int   fi0 = 0, fi1 = 0, fi2 = 0;
  for (int t = 0; t < 96; ++t) {
    float d2 = sdist[wave][t];
    int   id = sidx[wave][t];
    if (d2 < fd0)      { fd2 = fd1; fi2 = fi1; fd1 = fd0; fi1 = fi0; fd0 = d2; fi0 = id; }
    else if (d2 < fd1) { fd2 = fd1; fi2 = fi1; fd1 = d2;  fi1 = id; }
    else if (d2 < fd2) { fd2 = d2;  fi2 = id; }
  }
  float w0 = 1.0f / (sqrtf(fd0) + 1e-10f);
  float w1 = 1.0f / (sqrtf(fd1) + 1e-10f);
  float w2 = 1.0f / (sqrtf(fd2) + 1e-10f);
  const float wsum = 1.0f / (w0 + w1 + w2);
  w0 *= wsum; w1 *= wsum; w2 *= wsum;

  // blend: lane handles 16 contiguous channels, emit bf16 (32B vector store)
  const float* fb = feat + (size_t)b * N1_ * C_;
  const float* f0 = fb + (size_t)fi0 * C_;
  const float* f1 = fb + (size_t)fi1 * C_;
  const float* f2 = fb + (size_t)fi2 * C_;
  const int c0 = lane * 16;
  union { u16 h[16]; v4u q[2]; } o;
  #pragma unroll
  for (int t = 0; t < 16; t += 4) {
    v4f a = *(const v4f*)(f0 + c0 + t);
    v4f c = *(const v4f*)(f1 + c0 + t);
    v4f d = *(const v4f*)(f2 + c0 + t);
    #pragma unroll
    for (int e = 0; e < 4; ++e)
      o.h[t + e] = f2bf(w0 * a[e] + w1 * c[e] + w2 * d[e]);
  }
  v4u* dst = (v4u*)(X + (size_t)pt * C_ + c0);
  dst[0] = o.q[0];
  dst[1] = o.q[1];
}

// ---------------------------------------------------------------- fused MLP
// out = relu(X @ W1^T + b1) @ W2^T + b2, H tile (128x512 bf16) lives in LDS.
// 8 waves arranged 4(M) x 2(N); each stage loops 4 chunks of 64 cols.
// K-loop is manually software-pipelined: next k-step's 12 fragment loads are
// issued before the current step's 8 WMMAs -> partial s_wait_loadcnt overlap.
union FragA { v16bf v; v4u q[2]; };

__global__ __launch_bounds__(256) void fused_mlp_kernel(const u16* __restrict__ X,
                                                        const u16* __restrict__ W1,
                                                        const float* __restrict__ b1,
                                                        const u16* __restrict__ W2,
                                                        const float* __restrict__ b2,
                                                        float* __restrict__ out) {
  extern __shared__ u16 sH[];          // [128][HSTRIDE]

  const int lane = threadIdx.x & 31;
  const int wave = threadIdx.x >> 5;
  const int wm = wave & 3;             // 4 waves along M
  const int wn = wave >> 2;            // 2 waves along N
  const int mBlock = blockIdx.x * 128;
  const int row  = lane & 15;
  const int half = lane >> 4;
  const int crow = half * 8;           // C/D row offset for this half-wave
  const int mWave = wm * 32;           // wave's first row within the block tile

  // ---------------- stage 1: H = relu(X @ W1^T + b1) -> LDS bf16 ----------------
  // A (16x32 bf16) lane layout: half0 -> K{0..7,16..23}, half1 -> K{8..15,24..31}
  const u16* aB0 = X + (size_t)(mBlock + mWave + row) * C_ + half * 8;
  const u16* aB1 = aB0 + (size_t)16 * C_;

  #pragma unroll 1
  for (int c = 0; c < 4; ++c) {
    const int n0 = c * 128 + wn * 64;
    const u16* wB[4];
    #pragma unroll
    for (int j = 0; j < 4; ++j)
      wB[j] = W1 + (size_t)(n0 + j * 16 + row) * C_ + half * 8;

    v8f acc[2][4] = {};
    // prologue: load k = 0
    FragA a0c, a1c, bc[4];
    a0c.q[0] = *(const v4u*)(aB0);
    a0c.q[1] = *(const v4u*)(aB0 + 16);
    a1c.q[0] = *(const v4u*)(aB1);
    a1c.q[1] = *(const v4u*)(aB1 + 16);
    #pragma unroll
    for (int j = 0; j < 4; ++j) {
      bc[j].q[0] = *(const v4u*)(wB[j]);
      bc[j].q[1] = *(const v4u*)(wB[j] + 16);
    }
    #pragma unroll 2
    for (int k = 0; k < C_; k += 32) {
      const int kn = (k + 32) & (C_ - 1);       // last iter wraps to 0 (harmless)
      FragA a0n, a1n, bn[4];
      a0n.q[0] = *(const v4u*)(aB0 + kn);
      a0n.q[1] = *(const v4u*)(aB0 + kn + 16);
      a1n.q[0] = *(const v4u*)(aB1 + kn);
      a1n.q[1] = *(const v4u*)(aB1 + kn + 16);
      #pragma unroll
      for (int j = 0; j < 4; ++j) {
        bn[j].q[0] = *(const v4u*)(wB[j] + kn);
        bn[j].q[1] = *(const v4u*)(wB[j] + kn + 16);
      }
      __builtin_prefetch(aB0 + kn + 32, 0, 3);
      __builtin_prefetch(aB1 + kn + 32, 0, 3);
      #pragma unroll
      for (int j = 0; j < 4; ++j) {
        acc[0][j] = __builtin_amdgcn_wmma_f32_16x16x32_bf16(
            false, a0c.v, false, bc[j].v, (short)0, acc[0][j], false, false);
        acc[1][j] = __builtin_amdgcn_wmma_f32_16x16x32_bf16(
            false, a1c.v, false, bc[j].v, (short)0, acc[1][j], false, false);
      }
      a0c = a0n; a1c = a1n;
      #pragma unroll
      for (int j = 0; j < 4; ++j) bc[j] = bn[j];
    }
    // epilogue -> LDS (bf16 + relu + bias)
    #pragma unroll
    for (int j = 0; j < 4; ++j) {
      const int gn = n0 + j * 16 + row;
      const float bv = b1[gn];
      #pragma unroll
      for (int i = 0; i < 2; ++i) {
        #pragma unroll
        for (int r = 0; r < 8; ++r) {
          float v = acc[i][j][r] + bv;
          v = fmaxf(v, 0.0f);
          sH[(mWave + i * 16 + crow + r) * HSTRIDE + gn] = f2bf(v);
        }
      }
    }
  }
  __syncthreads();

  // ---------------- stage 2: out = H @ W2^T + b2 (A-fragments from LDS) ----------------
  const u16* hB0 = sH + (mWave + row) * HSTRIDE + half * 8;
  const u16* hB1 = hB0 + 16 * HSTRIDE;

  #pragma unroll 1
  for (int c = 0; c < 4; ++c) {
    const int n0 = c * 128 + wn * 64;
    const u16* wB[4];
    #pragma unroll
    for (int j = 0; j < 4; ++j)
      wB[j] = W2 + (size_t)(n0 + j * 16 + row) * C_ + half * 8;

    v8f acc[2][4] = {};
    FragA a0c, a1c, bc[4];
    a0c.q[0] = *(const v4u*)(hB0);            // ds_load_b128
    a0c.q[1] = *(const v4u*)(hB0 + 16);
    a1c.q[0] = *(const v4u*)(hB1);
    a1c.q[1] = *(const v4u*)(hB1 + 16);
    #pragma unroll
    for (int j = 0; j < 4; ++j) {
      bc[j].q[0] = *(const v4u*)(wB[j]);
      bc[j].q[1] = *(const v4u*)(wB[j] + 16);
    }
    #pragma unroll 2
    for (int k = 0; k < C_; k += 32) {
      const int kn = (k + 32) & (C_ - 1);
      FragA a0n, a1n, bn[4];
      a0n.q[0] = *(const v4u*)(hB0 + kn);
      a0n.q[1] = *(const v4u*)(hB0 + kn + 16);
      a1n.q[0] = *(const v4u*)(hB1 + kn);
      a1n.q[1] = *(const v4u*)(hB1 + kn + 16);
      #pragma unroll
      for (int j = 0; j < 4; ++j) {
        bn[j].q[0] = *(const v4u*)(wB[j] + kn);
        bn[j].q[1] = *(const v4u*)(wB[j] + kn + 16);
      }
      #pragma unroll
      for (int j = 0; j < 4; ++j) {
        acc[0][j] = __builtin_amdgcn_wmma_f32_16x16x32_bf16(
            false, a0c.v, false, bc[j].v, (short)0, acc[0][j], false, false);
        acc[1][j] = __builtin_amdgcn_wmma_f32_16x16x32_bf16(
            false, a1c.v, false, bc[j].v, (short)0, acc[1][j], false, false);
      }
      a0c = a0n; a1c = a1n;
      #pragma unroll
      for (int j = 0; j < 4; ++j) bc[j] = bn[j];
    }
    // epilogue -> global f32
    #pragma unroll
    for (int j = 0; j < 4; ++j) {
      const int gn = n0 + j * 16 + row;
      const float bv = b2[gn];
      #pragma unroll
      for (int i = 0; i < 2; ++i) {
        #pragma unroll
        for (int r = 0; r < 8; ++r) {
          const size_t o = (size_t)(mBlock + mWave + i * 16 + crow + r) * C_ + gn;
          out[o] = acc[i][j][r] + bv;
        }
      }
    }
  }
}

// ---------------------------------------------------------------- launch
extern "C" void kernel_launch(void* const* d_in, const int* in_sizes, int n_in,
                              void* d_out, int out_size, void* d_ws, size_t ws_size,
                              hipStream_t stream) {
  (void)in_sizes; (void)n_in; (void)out_size; (void)ws_size;
  const float* xyzd = (const float*)d_in[0];
  const float* xyzs = (const float*)d_in[1];
  const float* feat = (const float*)d_in[2];
  const float* W1   = (const float*)d_in[3];
  const float* b1   = (const float*)d_in[4];
  const float* W2   = (const float*)d_in[5];
  const float* b2   = (const float*)d_in[6];

  const size_t Mrows = (size_t)M_TOTAL;
  u16* Xbf  = (u16*)d_ws;                 // 64 MB
  u16* W1bf = Xbf + Mrows * C_;           // 0.5 MB
  u16* W2bf = W1bf + (size_t)C_ * C_;     // 0.5 MB

  cvt_bf16_kernel<<<(C_ * C_ + 255) / 256, 256, 0, stream>>>(W1, W1bf, C_ * C_);
  cvt_bf16_kernel<<<(C_ * C_ + 255) / 256, 256, 0, stream>>>(W2, W2bf, C_ * C_);

  interp_kernel<<<M_TOTAL / 8, 256, 0, stream>>>(xyzd, xyzs, feat, Xbf);

  fused_mlp_kernel<<<M_TOTAL / 128, 256, LDS_BYTES, stream>>>(Xbf, W1bf, b1, W2bf, b2,
                                                              (float*)d_out);
}
